// MoE_37263136260195
// MI455X (gfx1250) — matrix operand order repaired and checked by reading, not compiled, not run on hardware
//
#include <hip/hip_runtime.h>

#define T_TOK 8192
#define DIM   1024
#define NEXP  16
#define INTER 2048
#define MT1   (T_TOK / 64)          // max 64-row M tiles per expert
#define NPAIR (T_TOK * 2)           // total (token, expert) pairs
#define HCAP  (NPAIR + 64)          // h rows incl. padding slack

typedef __attribute__((ext_vector_type(16))) __bf16       v16bf;
typedef __attribute__((ext_vector_type(8)))  float        v8f;
typedef __attribute__((ext_vector_type(4)))  float        v4f;
typedef __attribute__((ext_vector_type(4)))  unsigned int v4u;

union AF { v16bf v; v4f q[2]; };
union BF { v16bf v; v4u q[2]; };
union PK { __bf16 b[8]; v4f q; };

__device__ inline unsigned int pack_bf2(float lo, float hi) {
  union { __bf16 b; unsigned short s; } a, c;
  a.b = (__bf16)lo; c.b = (__bf16)hi;
  return (unsigned int)a.s | ((unsigned int)c.s << 16);
}

// ---------------- init: zero y, counters, h slack rows ----------------
__global__ void moe_init_kernel(float* __restrict__ out, int* __restrict__ cnt,
                                __bf16* __restrict__ hslack) {
  size_t i = (size_t)blockIdx.x * blockDim.x + threadIdx.x;
  if (i < (size_t)T_TOK * DIM) out[i] = 0.0f;
  if (i < NEXP) cnt[i] = 0;
  if (i < (size_t)64 * INTER) hslack[i] = (__bf16)0.0f;
}

// ---------------- gate: one wave per token ----------------
__global__ void moe_gate_kernel(const float* __restrict__ x,
                                const float* __restrict__ Wg,
                                const float* __restrict__ bg,
                                float* __restrict__ wdense,
                                int* __restrict__ cnt) {
  int wave = threadIdx.x >> 5;
  int lane = threadIdx.x & 31;
  int t = blockIdx.x * 8 + wave;
  if (t >= T_TOK) return;

  float s[NEXP];
#pragma unroll
  for (int e = 0; e < NEXP; ++e) s[e] = 0.0f;

  const float* xr = x + (size_t)t * DIM;
  for (int d = lane; d < DIM; d += 32) {
    float xv = xr[d];
    const float* wr = Wg + (size_t)d * NEXP;
#pragma unroll
    for (int e = 0; e < NEXP; ++e) s[e] = fmaf(xv, wr[e], s[e]);
  }
#pragma unroll
  for (int e = 0; e < NEXP; ++e) {
    for (int off = 16; off > 0; off >>= 1) s[e] += __shfl_xor(s[e], off, 32);
    s[e] += bg[e];
  }
  if (lane == 0) {
    int i1 = 0; float m1 = s[0];
    for (int e = 1; e < NEXP; ++e) if (s[e] > m1) { m1 = s[e]; i1 = e; }
    int i2 = (i1 == 0) ? 1 : 0; float m2 = s[i2];
    for (int e = 0; e < NEXP; ++e)
      if (e != i1 && s[e] > m2) { m2 = s[e]; i2 = e; }
    // softmax top-2 renormalized: w1 = 1/(1+exp(m2-m1))
    float w2 = 1.0f / (1.0f + expf(m1 - m2));
    float w1 = 1.0f - w2;
    float* wrow = wdense + (size_t)t * NEXP;
#pragma unroll
    for (int e = 0; e < NEXP; ++e) wrow[e] = 0.0f;
    wrow[i1] = w1;
    atomicAdd(&cnt[i1], 1);
    if (w2 > 0.0f) { wrow[i2] = w2; atomicAdd(&cnt[i2], 1); }
  }
}

// ---------------- exclusive prefix over 16 counts ----------------
__global__ void moe_prefix_kernel(const int* __restrict__ cnt, int* __restrict__ offs) {
  if (threadIdx.x == 0) {
    int a = 0;
    for (int e = 0; e < NEXP; ++e) { offs[e] = a; a += cnt[e]; }
  }
}

// ---------------- deterministic compaction per expert ----------------
__global__ void moe_fill_kernel(const float* __restrict__ wdense,
                                const int* __restrict__ offs,
                                int* __restrict__ pairTok,
                                float* __restrict__ pairW) {
  __shared__ int sc[256];
  int e = blockIdx.x;
  int tid = threadIdx.x;
  int base = offs[e];
  for (int c = 0; c < T_TOK / 256; ++c) {
    int t = c * 256 + tid;
    float w = wdense[(size_t)t * NEXP + e];
    int flag = (w > 0.0f) ? 1 : 0;
    sc[tid] = flag;
    __syncthreads();
    for (int off = 1; off < 256; off <<= 1) {
      int v = (tid >= off) ? sc[tid - off] : 0;
      __syncthreads();
      sc[tid] += v;
      __syncthreads();
    }
    int incl = sc[tid];
    int total = sc[255];
    if (flag) {
      int pos = base + incl - 1;
      pairTok[pos] = t;
      pairW[pos] = w;
    }
    base += total;
    __syncthreads();
  }
}

// ---------------- GEMM1: h = silu(x@W1+b1) * (x@W3+b3), bf16 WMMA ----------------
__global__ void __launch_bounds__(256)
moe_gemm1_kernel(const float* __restrict__ x,
                 const float* __restrict__ W1, const float* __restrict__ b1,
                 const float* __restrict__ W3, const float* __restrict__ b3,
                 const int* __restrict__ offs, const int* __restrict__ cnt,
                 const int* __restrict__ pairTok,
                 __bf16* __restrict__ hbuf) {
  int e = blockIdx.x / MT1;
  int mt = blockIdx.x % MT1;
  int row_start = offs[e] + mt * 64;
  int row_end = offs[e] + cnt[e];
  if (row_start >= row_end) return;
  int n0 = blockIdx.y * 128;

  // double-buffered staging
  __shared__ __align__(16) __bf16       lA[2][64][40];    // [row][k], padded 80B rows
  __shared__ __align__(16) unsigned int lB1[2][128][20];  // fragment-major: [n][kpair]
  __shared__ __align__(16) unsigned int lB3[2][128][20];
  __shared__ int tok[64];

  int tid = threadIdx.x;
  if (tid < 64) {
    int p = row_start + tid;
    tok[tid] = pairTok[(p < row_end) ? p : row_start];
  }
  __syncthreads();

  const float* W1e = W1 + (size_t)e * DIM * INTER;
  const float* W3e = W3 + (size_t)e * DIM * INTER;

  int wave = tid >> 5, lane = tid & 31;
  int wm = wave >> 2, wn = wave & 3;

  int ar = tid >> 2;                 // A staging: row 0..63
  int ak = (tid & 3) * 8;            // k chunk within 32
  const float* axp = x + (size_t)tok[ar] * DIM + ak;
  int bkp = tid >> 4;                // B staging: k-pair 0..15
  int bn = (tid & 15) * 8;           // n chunk (8 cols)

  auto stage = [&](int buf, int kb) {
    // A tile: gather + f32->bf16, row-major
    v4f f0 = *(const v4f*)(axp + kb);
    v4f f1 = *(const v4f*)(axp + kb + 4);
    PK pk;
#pragma unroll
    for (int j = 0; j < 4; ++j) { pk.b[j] = (__bf16)f0[j]; pk.b[j + 4] = (__bf16)f1[j]; }
    *(v4f*)&lA[buf][ar][ak] = pk.q;
    // B tiles: k-pair packed dwords, fragment-major [n][kp]
    const float* r0 = W1e + (size_t)(kb + 2 * bkp) * INTER + n0 + bn;
    const float* r1 = r0 + INTER;
    v4f a0 = *(const v4f*)r0, a1 = *(const v4f*)(r0 + 4);
    v4f c0 = *(const v4f*)r1, c1 = *(const v4f*)(r1 + 4);
    const float* s0 = W3e + (size_t)(kb + 2 * bkp) * INTER + n0 + bn;
    const float* s1 = s0 + INTER;
    v4f d0 = *(const v4f*)s0, d1 = *(const v4f*)(s0 + 4);
    v4f g0 = *(const v4f*)s1, g1 = *(const v4f*)(s1 + 4);
    unsigned int* w1d = &lB1[buf][bn][bkp];
    unsigned int* w3d = &lB3[buf][bn][bkp];
#pragma unroll
    for (int j = 0; j < 4; ++j) {
      w1d[j * 20]       = pack_bf2(a0[j], c0[j]);
      w1d[(j + 4) * 20] = pack_bf2(a1[j], c1[j]);
      w3d[j * 20]       = pack_bf2(d0[j], g0[j]);
      w3d[(j + 4) * 20] = pack_bf2(d1[j], g1[j]);
    }
  };

  const v8f vzero = {0.f, 0.f, 0.f, 0.f, 0.f, 0.f, 0.f, 0.f};
  v8f acc1[2][2], acc3[2][2];
#pragma unroll
  for (int i = 0; i < 2; ++i)
#pragma unroll
    for (int j = 0; j < 2; ++j) { acc1[i][j] = vzero; acc3[i][j] = vzero; }

  stage(0, 0);
  const int nsteps = DIM / 32;
  for (int s = 0; s < nsteps; ++s) {
    __syncthreads();
    int cur = s & 1;
    if (s + 1 < nsteps) stage(cur ^ 1, (s + 1) * 32);

    AF a[2];
#pragma unroll
    for (int mi = 0; mi < 2; ++mi) {
      int row = wm * 32 + mi * 16 + (lane & 15);
      int ka = (lane >> 4) * 8;
      a[mi].q[0] = *(const v4f*)&lA[cur][row][ka];
      a[mi].q[1] = *(const v4f*)&lA[cur][row][ka + 16];
    }
    BF bf1[2], bf3[2];
#pragma unroll
    for (int ni = 0; ni < 2; ++ni) {
      int n = wn * 32 + ni * 16 + (lane & 15);
      int kpb = (lane >> 4) * 4;
      const v4u* p1 = (const v4u*)&lB1[cur][n][kpb];
      const v4u* p3 = (const v4u*)&lB3[cur][n][kpb];
      bf1[ni].q[0] = p1[0]; bf1[ni].q[1] = p1[2];
      bf3[ni].q[0] = p3[0]; bf3[ni].q[1] = p3[2];
    }
#pragma unroll
    for (int mi = 0; mi < 2; ++mi)
#pragma unroll
      for (int ni = 0; ni < 2; ++ni) {
        acc1[mi][ni] = __builtin_amdgcn_wmma_f32_16x16x32_bf16(
            false, a[mi].v, false, bf1[ni].v, (short)0, acc1[mi][ni], false, false);
        acc3[mi][ni] = __builtin_amdgcn_wmma_f32_16x16x32_bf16(
            false, a[mi].v, false, bf3[ni].v, (short)0, acc3[mi][ni], false, false);
      }
  }

  const float* b1e = b1 + (size_t)e * INTER;
  const float* b3e = b3 + (size_t)e * INTER;
#pragma unroll
  for (int mi = 0; mi < 2; ++mi)
#pragma unroll
    for (int ni = 0; ni < 2; ++ni) {
      int n = n0 + wn * 32 + ni * 16 + (lane & 15);
      float bb1 = b1e[n], bb3 = b3e[n];
#pragma unroll
      for (int r = 0; r < 8; ++r) {
        int m = wm * 32 + mi * 16 + (lane >> 4) * 8 + r;
        int p = row_start + m;
        if (p < row_end) {
          float z1 = acc1[mi][ni][r] + bb1;
          float z3 = acc3[mi][ni][r] + bb3;
          float hv = (z1 / (1.0f + __expf(-z1))) * z3;   // silu(z1) * z3
          hbuf[(size_t)p * INTER + n] = (__bf16)hv;
        }
      }
    }
}

// ---------------- GEMM2: y[t] += (h @ W2 + b2) * w, atomic scatter ----------------
__global__ void __launch_bounds__(256)
moe_gemm2_kernel(const __bf16* __restrict__ hbuf,
                 const float* __restrict__ W2, const float* __restrict__ b2,
                 const int* __restrict__ offs, const int* __restrict__ cnt,
                 const int* __restrict__ pairTok, const float* __restrict__ pairW,
                 float* __restrict__ out) {
  int e = blockIdx.x / MT1;
  int mt = blockIdx.x % MT1;
  int row_start = offs[e] + mt * 64;
  int row_end = offs[e] + cnt[e];
  if (row_start >= row_end) return;
  int n0 = blockIdx.y * 128;

  __shared__ __align__(16) unsigned int lB[2][128][20];  // fragment-major [n][kp]

  int tid = threadIdx.x, wave = tid >> 5, lane = tid & 31;
  int wm = wave >> 2, wn = wave & 3;
  const float* W2e = W2 + (size_t)e * INTER * DIM;

  int bkp = tid >> 4;
  int bn = (tid & 15) * 8;

  auto stage = [&](int buf, int kb) {
    const float* r0 = W2e + (size_t)(kb + 2 * bkp) * DIM + n0 + bn;
    const float* r1 = r0 + DIM;
    v4f a0 = *(const v4f*)r0, a1 = *(const v4f*)(r0 + 4);
    v4f c0 = *(const v4f*)r1, c1 = *(const v4f*)(r1 + 4);
    unsigned int* wd = &lB[buf][bn][bkp];
#pragma unroll
    for (int j = 0; j < 4; ++j) {
      wd[j * 20]       = pack_bf2(a0[j], c0[j]);
      wd[(j + 4) * 20] = pack_bf2(a1[j], c1[j]);
    }
  };

  const v8f vzero = {0.f, 0.f, 0.f, 0.f, 0.f, 0.f, 0.f, 0.f};
  v8f acc[2][2];
#pragma unroll
  for (int i = 0; i < 2; ++i)
#pragma unroll
    for (int j = 0; j < 2; ++j) acc[i][j] = vzero;

  stage(0, 0);
  const int nsteps = INTER / 32;
  for (int s = 0; s < nsteps; ++s) {
    __syncthreads();
    int cur = s & 1;
    int kb = s * 32;
    if (s + 1 < nsteps) stage(cur ^ 1, kb + 32);

    AF a[2];
#pragma unroll
    for (int mi = 0; mi < 2; ++mi) {
      int p = row_start + wm * 32 + mi * 16 + (lane & 15);   // slack rows are valid mem
      const __bf16* hp = hbuf + (size_t)p * INTER + kb + (lane >> 4) * 8;
      a[mi].q[0] = *(const v4f*)hp;
      a[mi].q[1] = *(const v4f*)(hp + 16);
    }
    BF bf[2];
#pragma unroll
    for (int ni = 0; ni < 2; ++ni) {
      int n = wn * 32 + ni * 16 + (lane & 15);
      int kpb = (lane >> 4) * 4;
      const v4u* p1 = (const v4u*)&lB[cur][n][kpb];
      bf[ni].q[0] = p1[0]; bf[ni].q[1] = p1[2];
    }
#pragma unroll
    for (int mi = 0; mi < 2; ++mi)
#pragma unroll
      for (int ni = 0; ni < 2; ++ni)
        acc[mi][ni] = __builtin_amdgcn_wmma_f32_16x16x32_bf16(
            false, a[mi].v, false, bf[ni].v, (short)0, acc[mi][ni], false, false);
  }

  const float* b2e = b2 + (size_t)e * DIM;
#pragma unroll
  for (int mi = 0; mi < 2; ++mi)
#pragma unroll
    for (int ni = 0; ni < 2; ++ni) {
      int n = n0 + wn * 32 + ni * 16 + (lane & 15);
      float bb = b2e[n];
#pragma unroll
      for (int r = 0; r < 8; ++r) {
        int m = wm * 32 + mi * 16 + (lane >> 4) * 8 + r;
        int p = row_start + m;
        if (p < row_end) {
          float w = pairW[p];
          int t = pairTok[p];
          atomicAdd(&out[(size_t)t * DIM + n], (acc[mi][ni][r] + bb) * w);
        }
      }
    }
}

// ---------------- host launcher ----------------
extern "C" void kernel_launch(void* const* d_in, const int* in_sizes, int n_in,
                              void* d_out, int out_size, void* d_ws, size_t ws_size,
                              hipStream_t stream) {
  const float* x  = (const float*)d_in[0];
  const float* Wg = (const float*)d_in[1];
  const float* bg = (const float*)d_in[2];
  const float* W1 = (const float*)d_in[3];
  const float* b1 = (const float*)d_in[4];
  const float* W2 = (const float*)d_in[5];
  const float* b2 = (const float*)d_in[6];
  const float* W3 = (const float*)d_in[7];
  const float* b3 = (const float*)d_in[8];
  float* out = (float*)d_out;

  char* ws = (char*)d_ws;
  size_t o = 0;
  auto alloc = [&](size_t bytes) -> char* {
    char* r = ws + o;
    o = (o + bytes + 255) & ~(size_t)255;
    return r;
  };
  int*    cnt     = (int*)alloc(NEXP * sizeof(int));
  int*    offs    = (int*)alloc(NEXP * sizeof(int));
  int*    pairTok = (int*)alloc(HCAP * sizeof(int));
  float*  pairW   = (float*)alloc(HCAP * sizeof(float));
  float*  wdense  = (float*)alloc((size_t)T_TOK * NEXP * sizeof(float));
  __bf16* hbuf    = (__bf16*)alloc((size_t)HCAP * INTER * sizeof(__bf16));

  moe_init_kernel<<<(T_TOK * DIM) / 256, 256, 0, stream>>>(
      out, cnt, hbuf + (size_t)NPAIR * INTER);
  moe_gate_kernel<<<T_TOK / 8, 256, 0, stream>>>(x, Wg, bg, wdense, cnt);
  moe_prefix_kernel<<<1, 32, 0, stream>>>(cnt, offs);
  moe_fill_kernel<<<NEXP, 256, 0, stream>>>(wdense, offs, pairTok, pairW);
  moe_gemm1_kernel<<<dim3(NEXP * MT1, INTER / 128), 256, 0, stream>>>(
      x, W1, b1, W3, b3, offs, cnt, pairTok, hbuf);
  moe_gemm2_kernel<<<dim3(NEXP * MT1, DIM / 128), 256, 0, stream>>>(
      hbuf, W2, b2, offs, cnt, pairTok, pairW, out);
}
